// MixHopConv_57561151701008
// MI455X (gfx1250) — matrix operand order repaired and checked by reading, not compiled, not run on hardware
//
#include <hip/hip_runtime.h>
#include <hip/hip_bf16.h>
#include <stdint.h>

// Problem constants (match reference)
#define Bn   16
#define Cc   64
#define Nn   512
#define Tt   64
#define OUTC 64
#define FDIM 192   // C*(DEPTH+1)

typedef __attribute__((ext_vector_type(16))) __bf16        v16bf;
typedef __attribute__((ext_vector_type(8)))  float         v8f;
typedef __attribute__((ext_vector_type(4)))  unsigned int  u32x4;

union Frag {
    v16bf v;
    u32x4 q[2];
};

// float -> bf16 (round to nearest even)
__device__ __forceinline__ unsigned short f2bf(float f) {
    union { float f; unsigned int u; } cv; cv.f = f;
    unsigned int u = cv.u;
    unsigned int r = u + 0x7FFFu + ((u >> 16) & 1u);
    return (unsigned short)(r >> 16);
}

// A fragment, 16x32 bf16: lane l<16 = row l, K in {h*8..h*8+7} U {h*8+16..h*8+23}
__device__ __forceinline__ v16bf ldA(const unsigned short* p) {
    Frag f; f.q[0] = *(const u32x4*)p; f.q[1] = *(const u32x4*)(p + 16); return f.v;
}
// B fragment, 32x16 bf16 via B^T row: lane l<16 = col l, K in {h*8..h*8+7} U {h*8+16..h*8+23}
__device__ __forceinline__ v16bf ldB(const unsigned short* p) {
    Frag f; f.q[0] = *(const u32x4*)p; f.q[1] = *(const u32x4*)(p + 16); return f.v;
}

__device__ __forceinline__ v8f wmma_bf16(v16bf a, v16bf b, v8f c) {
    return __builtin_amdgcn_wmma_f32_16x16x32_bf16(false, a, false, b, (short)0, c, false, false);
}

// ---------------------------------------------------------------------------
// Prep kernel 1: row-normalize adjacency, convert to bf16.
// ---------------------------------------------------------------------------
__global__ void __launch_bounds__(256)
norm_adj_kernel(const float* __restrict__ adj, unsigned short* __restrict__ normBf) {
    __shared__ float red[256];
    const int i   = blockIdx.x;
    const int tid = threadIdx.x;
    const float a0 = adj[i * Nn + tid];
    const float a1 = adj[i * Nn + tid + 256];
    red[tid] = a0 + a1;
    __syncthreads();
    for (int off = 128; off > 0; off >>= 1) {
        if (tid < off) red[tid] += red[tid + off];
        __syncthreads();
    }
    const float inv = 1.0f / (red[0] + 1e-6f);
    normBf[i * Nn + tid]       = f2bf(a0 * inv);
    normBf[i * Nn + tid + 256] = f2bf(a1 * inv);
}

// ---------------------------------------------------------------------------
// Prep kernel 2: W (OUTC x FDIM row-major f32) -> bf16
// ---------------------------------------------------------------------------
__global__ void __launch_bounds__(256)
convW_kernel(const float* __restrict__ W, unsigned short* __restrict__ Wbf) {
    const int i = blockIdx.x * 256 + threadIdx.x;
    if (i < OUTC * FDIM) Wbf[i] = f2bf(W[i]);
}

// ---------------------------------------------------------------------------
// Fused per-sample MixHop kernel. One workgroup (8 wave32) per sample bt.
// LDS (dynamic, 320KB): H0/H1 (64x512 bf16 channel-major), catT (512x192 bf16
// node-major). 2x2 register blocking: 4 fragment loads feed 4 WMMAs.
// ---------------------------------------------------------------------------
__global__ void __launch_bounds__(256)
mixhop_fused(const float* __restrict__ x,
             const unsigned short* __restrict__ normBf,   // [N][N] bf16 row-major
             const unsigned short* __restrict__ Wbf,      // [OUTC][FDIM] bf16
             const float* __restrict__ bias,              // [OUTC] f32
             float* __restrict__ out)                     // [B][OUTC][N][T] f32
{
    extern __shared__ unsigned short smem[];
    unsigned short* H0   = smem;                     // 64*512
    unsigned short* H1   = smem + Cc * Nn;           // 64*512
    unsigned short* catT = smem + 2 * Cc * Nn;       // 512*192

    const int s    = blockIdx.x;       // sample index = b*T + t
    const int bb   = s >> 6;           // / T
    const int tt   = s & (Tt - 1);     // % T
    const int tid  = threadIdx.x;
    const int wave = tid >> 5;
    const int lane = tid & 31;
    const int l16  = lane & 15;
    const int half = lane >> 4;

    // ---- Phase 0: gather x slice (C x N, stride T in n) -> H0 + catT sec 0
    for (int i = tid; i < Cc * Nn; i += 256) {
        const int c = i >> 9;          // / N
        const int n = i & (Nn - 1);    // % N
        const float vx = x[(((size_t)bb * Cc + c) * Nn + n) * Tt + tt];
        const unsigned short hv = f2bf(vx);
        H0[c * Nn + n]      = hv;
        catT[n * FDIM + c]  = hv;
    }
    __syncthreads();

    // ---- Phase 1/2: two propagation levels: Hl = H(l-1) @ norm^T
    for (int lev = 1; lev <= 2; ++lev) {
        const unsigned short* Asrc = (lev == 1) ? H0 : H1;
        unsigned short*       Hdst = (lev == 1) ? H1 : nullptr;

        // 32 macro-tiles: (2 m-tiles) x (2 n-tiles) each; 4 per wave
        for (int macro = wave; macro < 32; macro += 8) {
            const int mt = (macro >> 4) << 1;   // 0 or 2
            const int nt = (macro & 15) << 1;   // 0,2,..,30
            v8f acc00 = {}, acc01 = {}, acc10 = {}, acc11 = {};
            const unsigned short* A0 = Asrc + (mt * 16 + l16) * Nn;
            const unsigned short* A1 = A0 + 16 * Nn;
            const unsigned short* B0 = normBf + (size_t)(nt * 16 + l16) * Nn;
            const unsigned short* B1 = B0 + 16 * Nn;
            for (int kk = 0; kk < Nn; kk += 32) {
                const v16bf a0 = ldA(A0 + kk + half * 8);
                const v16bf a1 = ldA(A1 + kk + half * 8);
                const v16bf b0 = ldB(B0 + kk + half * 8);
                const v16bf b1 = ldB(B1 + kk + half * 8);
                acc00 = wmma_bf16(a0, b0, acc00);
                acc01 = wmma_bf16(a0, b1, acc01);
                acc10 = wmma_bf16(a1, b0, acc10);
                acc11 = wmma_bf16(a1, b1, acc11);
            }
            // D tiles: element (m = v + 8*half, n = l16)
#pragma unroll
            for (int mi = 0; mi < 2; ++mi) {
#pragma unroll
                for (int ni = 0; ni < 2; ++ni) {
                    const v8f acc = mi ? (ni ? acc11 : acc10) : (ni ? acc01 : acc00);
                    const int nglob = (nt + ni) * 16 + l16;
#pragma unroll
                    for (int v = 0; v < 8; ++v) {
                        const int m = (mt + mi) * 16 + v + 8 * half;
                        const unsigned short hv = f2bf(acc[v]);
                        if (Hdst) Hdst[m * Nn + nglob] = hv;
                        catT[nglob * FDIM + lev * Cc + m] = hv;
                    }
                }
            }
        }
        __syncthreads();
    }

    // ---- Phase 3: Y^T = W @ cat^T + bias ; K = 192, 2x2 blocked
    for (int macro = wave; macro < 32; macro += 8) {
        const int mt = (macro >> 4) << 1;
        const int nt = (macro & 15) << 1;
        v8f acc00 = {}, acc01 = {}, acc10 = {}, acc11 = {};
        const unsigned short* A0 = Wbf  + (mt * 16 + l16) * FDIM;   // W rows (global)
        const unsigned short* A1 = A0 + 16 * FDIM;
        const unsigned short* B0 = catT + (nt * 16 + l16) * FDIM;   // cat rows (LDS)
        const unsigned short* B1 = B0 + 16 * FDIM;
#pragma unroll
        for (int kk = 0; kk < FDIM; kk += 32) {
            const v16bf a0 = ldA(A0 + kk + half * 8);
            const v16bf a1 = ldA(A1 + kk + half * 8);
            const v16bf b0 = ldB(B0 + kk + half * 8);
            const v16bf b1 = ldB(B1 + kk + half * 8);
            acc00 = wmma_bf16(a0, b0, acc00);
            acc01 = wmma_bf16(a0, b1, acc01);
            acc10 = wmma_bf16(a1, b0, acc10);
            acc11 = wmma_bf16(a1, b1, acc11);
        }
#pragma unroll
        for (int mi = 0; mi < 2; ++mi) {
#pragma unroll
            for (int ni = 0; ni < 2; ++ni) {
                const v8f acc = mi ? (ni ? acc11 : acc10) : (ni ? acc01 : acc00);
                const int nglob = (nt + ni) * 16 + l16;
#pragma unroll
                for (int v = 0; v < 8; ++v) {
                    const int o = (mt + mi) * 16 + v + 8 * half;
                    const float y = acc[v] + bias[o];
                    out[(((size_t)bb * OUTC + o) * Nn + nglob) * Tt + tt] = y;
                }
            }
        }
    }
}

// ---------------------------------------------------------------------------
extern "C" void kernel_launch(void* const* d_in, const int* in_sizes, int n_in,
                              void* d_out, int out_size, void* d_ws, size_t ws_size,
                              hipStream_t stream) {
    const float* x   = (const float*)d_in[0];   // (B, C, N, T)
    const float* adj = (const float*)d_in[1];   // (N, N)
    const float* W   = (const float*)d_in[2];   // (OUTC, FDIM)
    const float* b   = (const float*)d_in[3];   // (OUTC,)
    float* out = (float*)d_out;

    unsigned short* normBf = (unsigned short*)d_ws;                       // 512 KB
    unsigned short* Wbf    = (unsigned short*)((char*)d_ws + (size_t)Nn * Nn * 2);

    norm_adj_kernel<<<Nn, 256, 0, stream>>>(adj, normBf);
    convW_kernel<<<(OUTC * FDIM + 255) / 256, 256, 0, stream>>>(W, Wbf);

    const int lds_bytes = (2 * Cc * Nn + Nn * FDIM) * 2;   // 320 KB
    (void)hipFuncSetAttribute(reinterpret_cast<const void*>(mixhop_fused),
                              hipFuncAttributeMaxDynamicSharedMemorySize, lds_bytes);
    mixhop_fused<<<Bn * Tt, 256, lds_bytes, stream>>>(x, normBf, Wbf, b, out);
}
